// NIN_54468775248387
// MI455X (gfx1250) — 8-bit matrix operand order repaired and checked by reading, compiled, not run on hardware
//
#include <hip/hip_runtime.h>

#define EPS 1e-4f

typedef __attribute__((ext_vector_type(8))) int   v8i;
typedef __attribute__((ext_vector_type(8))) float v8f;
typedef __attribute__((ext_vector_type(2))) float v2f;

__device__ __forceinline__ signed char sgnf(float x) {
  return x > 0.f ? (signed char)1 : (x < 0.f ? (signed char)-1 : (signed char)0);
}

__global__ void alpha_kernel(const float* __restrict__ W, int K, float* __restrict__ alpha)
{
  __shared__ float red[256];
  const float* w = W + (long)blockIdx.x * K;
  float s = 0.f;
  for (int k = threadIdx.x; k < K; k += 256) s += fabsf(w[k]);
  red[threadIdx.x] = s;
  __syncthreads();
  for (int o = 128; o > 0; o >>= 1) {
    if ((int)threadIdx.x < o) red[threadIdx.x] += red[threadIdx.x + o];
    __syncthreads();
  }
  if (threadIdx.x == 0) alpha[blockIdx.x] = red[0] / (float)K;
}

__global__ void thresh_kernel(const float* __restrict__ bm, const float* __restrict__ bv,
                              const float* __restrict__ bg, const float* __restrict__ bb,
                              float* __restrict__ thr, int C)
{
  int c = blockIdx.x * blockDim.x + threadIdx.x;
  if (c >= C) return;
  float g = fmaxf(bg[c], 0.01f);
  thr[c] = bm[c] - bb[c] * sqrtf(bv[c] + EPS) / g;
}

__global__ void pack_b_kernel(const float* __restrict__ W, int O, int K, int nKT,
                              unsigned int* __restrict__ Bpack, int totalDwords)
{
  int idx = blockIdx.x * blockDim.x + threadIdx.x;
  if (idx >= totalDwords) return;
  int i = idx & 7;
  int lane = (idx >> 3) & 31;
  int tile = idx >> 8;
  int kt = tile % nKT;
  int nt = tile / nKT;
  int n = nt * 16 + (lane & 15);
  int kb = kt * 64 + ((lane >> 4) * 8) + (i >> 1) * 16 + (i & 1) * 4;
  unsigned int packed = 0;
  for (int j = 0; j < 4; ++j) {
    int k = kb + j;
    signed char b = 0;
    if (n < O && k < K) b = sgnf(W[(long)n * K + k]);
    packed |= ((unsigned int)(unsigned char)b) << (8 * j);
  }
  Bpack[idx] = packed;
}

template <int KH, int KW>
__global__ void pack_a_kernel(const float* __restrict__ act, const float* __restrict__ thr,
                              int C, int wShift, int pad, int K, int nKT,
                              unsigned int* __restrict__ Apack)
{
  const int i = threadIdx.x & 7;
  const int lane = threadIdx.x >> 3;
  const int mt = blockIdx.x;
  const int kt = blockIdx.y;
  const int W = 1 << wShift;
  const int spShift = 2 * wShift;
  int m = mt * 16 + (lane & 15);
  int k0 = kt * 64 + ((lane >> 4) * 8) + ((i >> 1) * 16) + ((i & 1) * 4);
  int nimg = m >> spShift;
  int s = m & ((1 << spShift) - 1);
  int oh = s >> wShift;
  int ow = s & (W - 1);
  unsigned int packed = 0;
#pragma unroll
  for (int j = 0; j < 4; ++j) {
    int k = k0 + j;
    signed char b = 0;
    if (k < K) {
      int c, kh, kw;
      if (KH == 1 && KW == 1) { c = k; kh = 0; kw = 0; }
      else {
        c = k / (KH * KW);
        int r = k - c * (KH * KW);
        kh = r / KW;
        kw = r - kh * KW;
      }
      int ih = oh - pad + kh;
      int iw = ow - pad + kw;
      if ((unsigned)ih < (unsigned)W && (unsigned)iw < (unsigned)W) {
        float x = act[(((long)nimg * C + c) << spShift) + (ih << wShift) + iw];
        float t = thr[c];
        b = x > t ? (signed char)1 : (x < t ? (signed char)-1 : (signed char)0);
      }
    }
    packed |= ((unsigned int)(unsigned char)b) << (8 * j);
  }
  Apack[(((long)(mt * nKT + kt)) << 8) + threadIdx.x] = packed;
}

template <int NT>
__global__ void __launch_bounds__(128) xnor_gemm_kernel(
    const int4* __restrict__ Apack, const int4* __restrict__ Bpack,
    const float* __restrict__ alpha, float* __restrict__ out,
    int nKT, int O, int spShift)
{
  __shared__ int4 shB[2][NT * 64];
  const int lane = threadIdx.x & 31;
  const int wave = threadIdx.x >> 5;
  const int mt = blockIdx.x * 4 + wave;
  const int nt0 = blockIdx.y * NT;
  v8i acc[NT] = {};
  const long aBase = ((long)mt * nKT) * 32 + lane;
#pragma unroll
  for (int it = 0; it < NT / 2; ++it) {
    int q = threadIdx.x + it * 128;
    shB[0][q] = Bpack[((long)(nt0 + (q >> 6)) * nKT) * 64 + (q & 63)];
  }
  __syncthreads();
  for (int kt = 0; kt < nKT; ++kt) {
    const int4* ap = Apack + (aBase + (long)kt * 32) * 2;
    int4 a0 = ap[0], a1 = ap[1];
    __builtin_prefetch((const void*)(ap + 64), 0, 1);
    v8i A;
    A[0] = a0.x; A[1] = a0.y; A[2] = a0.z; A[3] = a0.w;
    A[4] = a1.x; A[5] = a1.y; A[6] = a1.z; A[7] = a1.w;
    const int4* bb = shB[kt & 1];
#pragma unroll
    for (int j = 0; j < NT; ++j) {
      int4 b0 = bb[j * 64 + lane * 2];
      int4 b1 = bb[j * 64 + lane * 2 + 1];
      v8i B;
      B[0] = b0.x; B[1] = b0.y; B[2] = b0.z; B[3] = b0.w;
      B[4] = b1.x; B[5] = b1.y; B[6] = b1.z; B[7] = b1.w;
      acc[j] = __builtin_amdgcn_wmma_i32_16x16x64_iu8(true, A, true, B, acc[j], false, false);
    }
    {
      int4* dst = shB[(kt + 1) & 1];
#pragma unroll
      for (int it = 0; it < NT / 2; ++it) {
        int q = threadIdx.x + it * 128;
        dst[q] = Bpack[((long)(nt0 + (q >> 6)) * nKT + (kt + 1)) * 64 + (q & 63)];
      }
    }
    __syncthreads();
  }
  const int SP = 1 << spShift;
  int mbase = mt * 16 + ((lane >> 4) << 3);
  int nimg = mbase >> spShift;
  int s0 = mbase & (SP - 1);
#pragma unroll
  for (int j = 0; j < NT; ++j) {
    int n = (nt0 + j) * 16 + (lane & 15);
    if (n >= O) continue;
    float al = alpha[n];
    float4 lo, hi;
    lo.x = fmaxf((float)acc[j][0] * al, 0.f);
    lo.y = fmaxf((float)acc[j][1] * al, 0.f);
    lo.z = fmaxf((float)acc[j][2] * al, 0.f);
    lo.w = fmaxf((float)acc[j][3] * al, 0.f);
    hi.x = fmaxf((float)acc[j][4] * al, 0.f);
    hi.y = fmaxf((float)acc[j][5] * al, 0.f);
    hi.z = fmaxf((float)acc[j][6] * al, 0.f);
    hi.w = fmaxf((float)acc[j][7] * al, 0.f);
    float* dst = out + (((long)nimg * O + n) << spShift) + s0;
    *(float4*)dst = lo;
    *(float4*)(dst + 4) = hi;
  }
}

__global__ void pack_a_conv1(const float* __restrict__ x, float* __restrict__ Af, int totalElems)
{
  int idx = blockIdx.x * blockDim.x + threadIdx.x;
  if (idx >= totalElems) return;
  int i = idx & 1;
  int lane = (idx >> 1) & 31;
  int tile = idx >> 6;
  int kt = tile % 19;
  int mt = tile / 19;
  int m = mt * 16 + (lane & 15);
  int k = kt * 4 + ((lane >> 4) << 1) + i;
  float v = 0.f;
  if (k < 75) {
    int c = k / 25; int r = k - c * 25; int kh = r / 5; int kw = r - kh * 5;
    int nimg = m >> 10;
    int s = m & 1023;
    int oh = s >> 5, ow = s & 31;
    int ih = oh - 2 + kh, iw = ow - 2 + kw;
    if ((unsigned)ih < 32u && (unsigned)iw < 32u)
      v = x[(((long)nimg * 3 + c) * 32 + ih) * 32 + iw];
  }
  Af[idx] = v;
}

__global__ void pack_b_conv1(const float* __restrict__ w, float* __restrict__ Bf)
{
  int idx = blockIdx.x * blockDim.x + threadIdx.x;
  if (idx >= 12 * 19 * 64) return;
  int i = idx & 1;
  int lane = (idx >> 1) & 31;
  int tile = idx >> 6;
  int kt = tile % 19;
  int nt = tile / 19;
  int n = nt * 16 + (lane & 15);
  int k = kt * 4 + ((lane >> 4) << 1) + i;
  Bf[idx] = (k < 75) ? w[n * 75 + k] : 0.f;
}

__global__ void __launch_bounds__(128) conv1_gemm_kernel(
    const float* __restrict__ Af, const float2* __restrict__ Bf2,
    const float* __restrict__ bias, const float* __restrict__ bm,
    const float* __restrict__ bv, float* __restrict__ out)
{
  __shared__ float2 shB[2][12 * 32];
  const int lane = threadIdx.x & 31;
  const int wave = threadIdx.x >> 5;
  const int mt = blockIdx.x * 4 + wave;
  v8f acc[12] = {};
#pragma unroll
  for (int it = 0; it < 3; ++it) {
    int q = threadIdx.x + it * 128;
    shB[0][q] = Bf2[((q >> 5) * 19) * 32 + (q & 31)];
  }
  __syncthreads();
  for (int kt = 0; kt < 19; ++kt) {
    const float* ap = Af + ((long)(mt * 19 + kt) * 32 + lane) * 2;
    v2f A; A[0] = ap[0]; A[1] = ap[1];
    const float2* bb = shB[kt & 1];
#pragma unroll
    for (int j = 0; j < 12; ++j) {
      float2 bv2 = bb[j * 32 + lane];
      v2f B; B[0] = bv2.x; B[1] = bv2.y;
      acc[j] = __builtin_amdgcn_wmma_f32_16x16x4_f32(false, A, false, B,
                                                     (short)0, acc[j], false, false);
    }
    {
      float2* dst = shB[(kt + 1) & 1];
#pragma unroll
      for (int it = 0; it < 3; ++it) {
        int q = threadIdx.x + it * 128;
        dst[q] = Bf2[((q >> 5) * 19 + (kt + 1)) * 32 + (q & 31)];
      }
    }
    __syncthreads();
  }
  int mbase = mt * 16 + ((lane >> 4) << 3);
  int nimg = mbase >> 10;
  int s0 = mbase & 1023;
#pragma unroll
  for (int j = 0; j < 12; ++j) {
    int n = j * 16 + (lane & 15);
    float b = bias[n];
    float mu = bm[n];
    float iv = rsqrtf(bv[n] + EPS);
    float4 lo, hi;
    lo.x = fmaxf((acc[j][0] + b - mu) * iv, 0.f);
    lo.y = fmaxf((acc[j][1] + b - mu) * iv, 0.f);
    lo.z = fmaxf((acc[j][2] + b - mu) * iv, 0.f);
    lo.w = fmaxf((acc[j][3] + b - mu) * iv, 0.f);
    hi.x = fmaxf((acc[j][4] + b - mu) * iv, 0.f);
    hi.y = fmaxf((acc[j][5] + b - mu) * iv, 0.f);
    hi.z = fmaxf((acc[j][6] + b - mu) * iv, 0.f);
    hi.w = fmaxf((acc[j][7] + b - mu) * iv, 0.f);
    float* dst = out + (((long)nimg * 192 + n) << 10) + s0;
    *(float4*)dst = lo;
    *(float4*)(dst + 4) = hi;
  }
}

__global__ void maxpool_3s2p1(const float* __restrict__ in, float* __restrict__ out,
                              int C, int IH, int IW, int OH, int OW)
{
  long idx = (long)blockIdx.x * blockDim.x + threadIdx.x;
  long total = 256L * C * OH * OW;
  if (idx >= total) return;
  int ow = (int)(idx % OW); long t = idx / OW;
  int oh = (int)(t % OH); t /= OH;
  int c = (int)(t % C); int n = (int)(t / C);
  const float* p = in + ((long)n * C + c) * IH * IW;
  float m = -3.4e38f;
  for (int kh = 0; kh < 3; ++kh) {
    int ih = oh * 2 - 1 + kh;
    if ((unsigned)ih >= (unsigned)IH) continue;
    for (int kw = 0; kw < 3; ++kw) {
      int iw = ow * 2 - 1 + kw;
      if ((unsigned)iw >= (unsigned)IW) continue;
      float v = p[ih * IW + iw];
      m = v > m ? v : m;
    }
  }
  out[idx] = m;
}

__global__ void avgpool_3s2p1(const float* __restrict__ in, float* __restrict__ out,
                              int C, int IH, int IW, int OH, int OW)
{
  long idx = (long)blockIdx.x * blockDim.x + threadIdx.x;
  long total = 256L * C * OH * OW;
  if (idx >= total) return;
  int ow = (int)(idx % OW); long t = idx / OW;
  int oh = (int)(t % OH); t /= OH;
  int c = (int)(t % C); int n = (int)(t / C);
  const float* p = in + ((long)n * C + c) * IH * IW;
  float s = 0.f;
  for (int kh = 0; kh < 3; ++kh) {
    int ih = oh * 2 - 1 + kh;
    if ((unsigned)ih >= (unsigned)IH) continue;
    for (int kw = 0; kw < 3; ++kw) {
      int iw = ow * 2 - 1 + kw;
      if ((unsigned)iw >= (unsigned)IW) continue;
      s += p[ih * IW + iw];
    }
  }
  out[idx] = s * (1.f / 9.f);
}

__global__ void conv5_kernel(const float* __restrict__ act, const float* __restrict__ bm,
                             const float* __restrict__ bv, const float* __restrict__ w,
                             const float* __restrict__ bias, float* __restrict__ tmp)
{
  int idx = blockIdx.x * blockDim.x + threadIdx.x;
  if (idx >= 256 * 10 * 64) return;
  int s = idx & 63; int t = idx >> 6;
  int o = t % 10; int n = t / 10;
  float sum = bias[o];
  for (int c = 0; c < 192; ++c) {
    float xn = (act[(((long)n * 192 + c) << 6) + s] - bm[c]) * rsqrtf(bv[c] + EPS);
    sum += xn * w[o * 192 + c];
  }
  tmp[idx] = sum > 0.f ? sum : 0.f;
}

__global__ void final_mean_kernel(const float* __restrict__ tmp, float* __restrict__ out)
{
  int idx = blockIdx.x * blockDim.x + threadIdx.x;
  if (idx >= 2560) return;
  const float* p = tmp + (long)idx * 64;
  float s = 0.f;
  for (int i = 0; i < 64; ++i) s += p[i];
  out[idx] = s * (1.f / 64.f);
}

extern "C" void kernel_launch(void* const* d_in, const int* in_sizes, int n_in,
                              void* d_out, int out_size, void* d_ws, size_t ws_size,
                              hipStream_t stream)
{
  (void)in_sizes; (void)n_in; (void)out_size; (void)ws_size;
  auto F = [&](int i) { return (const float*)d_in[i]; };

  char* ws = (char*)d_ws;
  size_t off = 0;
  auto alloc = [&](size_t bytes) -> char* {
    char* p = ws + off; off += (bytes + 255) & ~(size_t)255; return p;
  };
  float* ACT0 = (float*)alloc(256UL * 192 * 32 * 32 * 4);
  float* ACT1 = (float*)alloc(256UL * 192 * 32 * 32 * 4);
  unsigned int* APACK = (unsigned int*)alloc(4096UL * 38 * 1024);
  unsigned int* B21 = (unsigned int*)alloc(12UL * 3 * 1024 + 1024);
  unsigned int* B22 = (unsigned int*)alloc(6UL * 3 * 1024 + 1024);
  unsigned int* B31 = (unsigned int*)alloc(12UL * 38 * 1024 + 1024);
  unsigned int* B32 = (unsigned int*)alloc(12UL * 3 * 1024 + 1024);
  unsigned int* B33 = (unsigned int*)alloc(12UL * 3 * 1024 + 1024);
  unsigned int* B41 = (unsigned int*)alloc(12UL * 27 * 1024 + 1024);
  unsigned int* B42 = (unsigned int*)alloc(12UL * 3 * 1024 + 1024);
  float* A21 = (float*)alloc(160 * 4);
  float* A22 = (float*)alloc(96 * 4);
  float* A31 = (float*)alloc(192 * 4);
  float* A32 = (float*)alloc(192 * 4);
  float* A33 = (float*)alloc(192 * 4);
  float* A41 = (float*)alloc(192 * 4);
  float* A42 = (float*)alloc(192 * 4);
  float* T21 = (float*)alloc(192 * 4);
  float* T22 = (float*)alloc(160 * 4);
  float* T31 = (float*)alloc(96 * 4);
  float* T32 = (float*)alloc(192 * 4);
  float* T33 = (float*)alloc(192 * 4);
  float* T41 = (float*)alloc(192 * 4);
  float* T42 = (float*)alloc(192 * 4);
  float* BC1 = (float*)alloc(12UL * 19 * 64 * 4 + 256);
  float* TMP5 = (float*)alloc(256UL * 10 * 64 * 4);

  auto prep_w = [&](const float* W, int O, int K, int NT, float* alpha, unsigned int* Bp) {
    int nKT = (K + 63) / 64;
    int Ntiles = (O + 15) / 16;
    int NtPad = ((Ntiles + NT - 1) / NT) * NT;
    alpha_kernel<<<O, 256, 0, stream>>>(W, K, alpha);
    int tot = NtPad * nKT * 256;
    pack_b_kernel<<<(tot + 255) / 256, 256, 0, stream>>>(W, O, K, nKT, Bp, tot);
  };
  prep_w(F(9),  160, 192,     12, A21, B21);
  prep_w(F(14),  96, 160,      6, A22, B22);
  prep_w(F(19), 192, 96 * 25, 12, A31, B31);
  prep_w(F(24), 192, 192,     12, A32, B32);
  prep_w(F(29), 192, 192,     12, A33, B33);
  prep_w(F(34), 192, 192 * 9, 12, A41, B41);
  prep_w(F(39), 192, 192,     12, A42, B42);

  auto prep_t = [&](int bnBase, float* thr, int C) {
    thresh_kernel<<<(C + 255) / 256, 256, 0, stream>>>(
        F(bnBase), F(bnBase + 1), F(bnBase + 2), F(bnBase + 3), thr, C);
  };
  prep_t(5, T21, 192); prep_t(10, T22, 160); prep_t(15, T31, 96);
  prep_t(20, T32, 192); prep_t(25, T33, 192); prep_t(30, T41, 192); prep_t(35, T42, 192);

  {
    int totA = 16384 * 19 * 64;
    pack_a_conv1<<<(totA + 255) / 256, 256, 0, stream>>>(F(0), (float*)APACK, totA);
    pack_b_conv1<<<(12 * 19 * 64 + 255) / 256, 256, 0, stream>>>(F(1), BC1);
    conv1_gemm_kernel<<<dim3(16384 / 4, 1), 128, 0, stream>>>(
        (const float*)APACK, (const float2*)BC1, F(2), F(3), F(4), ACT0);
  }

  auto run_xnor = [&](const float* act_in, float* act_out,
                      const float* thr, const unsigned int* Bp, const float* alpha,
                      int Cin, int Cout, int KH, int pad, int wShift, int NT) {
    int Hs = 1 << wShift;
    int SP = Hs * Hs;
    int Mtiles = (256 * SP) / 16;
    int K = Cin * KH * KH;
    int nKT = (K + 63) / 64;
    int Ntiles = (Cout + 15) / 16;
    int NtPad = ((Ntiles + NT - 1) / NT) * NT;
    dim3 pgrid(Mtiles, nKT);
    if (KH == 1)
      pack_a_kernel<1, 1><<<pgrid, 256, 0, stream>>>(act_in, thr, Cin, wShift, pad, K, nKT, APACK);
    else if (KH == 3)
      pack_a_kernel<3, 3><<<pgrid, 256, 0, stream>>>(act_in, thr, Cin, wShift, pad, K, nKT, APACK);
    else
      pack_a_kernel<5, 5><<<pgrid, 256, 0, stream>>>(act_in, thr, Cin, wShift, pad, K, nKT, APACK);
    dim3 grid(Mtiles / 4, NtPad / NT);
    if (NT == 12) {
      xnor_gemm_kernel<12><<<grid, 128, 0, stream>>>(
          (const int4*)APACK, (const int4*)Bp, alpha, act_out, nKT, Cout, 2 * wShift);
    } else {
      xnor_gemm_kernel<6><<<grid, 128, 0, stream>>>(
          (const int4*)APACK, (const int4*)Bp, alpha, act_out, nKT, Cout, 2 * wShift);
    }
  };

  run_xnor(ACT0, ACT1, T21, B21, A21, 192, 160, 1, 0, 5, 12);
  run_xnor(ACT1, ACT0, T22, B22, A22, 160,  96, 1, 0, 5, 6);
  {
    long tot = 256L * 96 * 16 * 16;
    maxpool_3s2p1<<<(unsigned)((tot + 255) / 256), 256, 0, stream>>>(ACT0, ACT1, 96, 32, 32, 16, 16);
  }
  run_xnor(ACT1, ACT0, T31, B31, A31,  96, 192, 5, 2, 4, 12);
  run_xnor(ACT0, ACT1, T32, B32, A32, 192, 192, 1, 0, 4, 12);
  run_xnor(ACT1, ACT0, T33, B33, A33, 192, 192, 1, 0, 4, 12);
  {
    long tot = 256L * 192 * 8 * 8;
    avgpool_3s2p1<<<(unsigned)((tot + 255) / 256), 256, 0, stream>>>(ACT0, ACT1, 192, 16, 16, 8, 8);
  }
  run_xnor(ACT1, ACT0, T41, B41, A41, 192, 192, 3, 1, 3, 12);
  run_xnor(ACT0, ACT1, T42, B42, A42, 192, 192, 1, 0, 3, 12);

  conv5_kernel<<<(256 * 10 * 64 + 255) / 256, 256, 0, stream>>>(
      ACT1, F(40), F(41), F(42), F(43), TMP5);
  final_mean_kernel<<<(2560 + 255) / 256, 256, 0, stream>>>(TMP5, (float*)d_out);
}
